// GraphEncoder_66735201845553
// MI455X (gfx1250) — matrix operand order repaired and checked by reading, not compiled, not run on hardware
//
#include <hip/hip_runtime.h>
#include <hip/hip_bf16.h>

// ---------------------------------------------------------------------------
// GraphEncoder for MI455X (gfx1250, wave32, WMMA).
// All dense GEMMs run on v_wmma_f32_16x16x32_f16 (f16 inputs, f32 accum).
// Main GEMM: A tile converted f32->f16 once per block into a double-buffered
// LDS tile (cuts conversion work 4x and removes the WMMA->VALU WAR hazard on
// the fragment registers); wave tile 16x32 with two independent accumulators.
// Weights are pre-converted to f16 [N][K] so B fragments are contiguous 32B
// global loads.  Scatter/softmax/segment ops are VALU/LDS kernels.
// ---------------------------------------------------------------------------

#define NN   2048    // nodes
#define DM   512     // d_model
#define NB   8       // graphs
#define PN   256     // nodes per graph
#define NL   3       // layers
#define NH   4       // heads
#define HD   128     // head dim

typedef __attribute__((ext_vector_type(16))) _Float16 v16h;
typedef __attribute__((ext_vector_type(8)))  _Float16 v8h;
typedef __attribute__((ext_vector_type(8)))  float    v8f;

// ------------------------------ WMMA GEMM ----------------------------------
// C[M,N] = act(A[M,K](f32) x Wt[N,K](f16) + bias), f32 out.
// Block = 256 threads = 8 waves; block tile 32(M) x 128(N); wave tile 16x32
// (2 accumulators).  N must be a multiple of 128, K a multiple of 32.
__global__ __launch_bounds__(256) void k_gemm_wmma(
    const float* __restrict__ A, const _Float16* __restrict__ Wt,
    const float* __restrict__ bias, float* __restrict__ C,
    int M, int N, int K, int lda, int ldb, int ldc, int relu)
{
    __shared__ _Float16 sa[2][32 * 32];    // double-buffered f16 A tile

    int tid  = threadIdx.x;
    int lane = tid & 31;
    int wave = tid >> 5;
    int m0     = blockIdx.x * 32;
    int m_base = m0 + (wave & 1) * 16;
    int n_base = blockIdx.y * 128 + (wave >> 1) * 32;

    // cooperative A staging: 8 threads per row, 4 floats each
    int sr = tid >> 3;                     // tile row 0..31
    int sc = (tid & 7) * 4;                // tile col 0..28
    int arow = m0 + sr; if (arow >= M) arow = M - 1;   // clamp (stores masked)
    const float* Aseg = A + (size_t)arow * lda + sc;

    // per-wave fragment addressing
    int fr = (wave & 1) * 16 + (lane & 15);            // row inside 32-row tile
    int lo = (lane >> 4) * 16;                         // A K sub-offset
    int ncol  = n_base + (lane & 15);
    int koffB = (lane >> 4) * 16;                      // B K sub-offset
    const _Float16* Brow0 = Wt + (size_t)ncol * ldb + koffB;
    const _Float16* Brow1 = Brow0 + (size_t)16 * ldb;

    auto fill = [&](int p, int k) {
        float4 v = *(const float4*)(Aseg + k);
        _Float16* d = &sa[p][sr * 32 + sc];
        d[0] = (_Float16)v.x; d[1] = (_Float16)v.y;
        d[2] = (_Float16)v.z; d[3] = (_Float16)v.w;
    };

    v8f acc0 = {}, acc1 = {};
    fill(0, 0);
    __syncthreads();
    int p = 0;
    for (int k = 0; k < K; k += 32) {
        if (k + 32 < K) {
            __builtin_prefetch(Aseg + k + 64);         // global_prefetch_b8
            __builtin_prefetch(Brow0 + k + 64);
            fill(p ^ 1, k + 32);                       // write other buffer
        }
        const _Float16* pa = &sa[p][fr * 32 + lo];
        v8h alo = *(const v8h*)(pa);                   // ds_load_b128
        v8h ahi = *(const v8h*)(pa + 8);               // ds_load_b128
        v16h a;
        #pragma unroll
        for (int i = 0; i < 8; ++i) { a[i] = alo[i]; a[8 + i] = ahi[i]; }
        v16h b0 = *(const v16h*)(Brow0 + k);           // 32B global loads
        v16h b1 = *(const v16h*)(Brow1 + k);
        acc0 = __builtin_amdgcn_wmma_f32_16x16x32_f16(
                   false, a, false, b0, (short)0, acc0, false, false);
        acc1 = __builtin_amdgcn_wmma_f32_16x16x32_f16(
                   false, a, false, b1, (short)0, acc1, false, false);
        __syncthreads();                               // fill done + reads done
        p ^= 1;
    }

    int mo = (lane >> 4) * 8;
    float bv0 = bias ? bias[ncol] : 0.0f;
    float bv1 = bias ? bias[ncol + 16] : 0.0f;
    #pragma unroll
    for (int r = 0; r < 8; ++r) {
        int m = m_base + mo + r;
        if (m < M) {
            float v0 = acc0[r] + bv0;
            float v1 = acc1[r] + bv1;
            if (relu) { v0 = fmaxf(v0, 0.0f); v1 = fmaxf(v1, 0.0f); }
            C[(size_t)m * ldc + ncol]      = v0;
            C[(size_t)m * ldc + ncol + 16] = v1;
        }
    }
}

// ------------------------- attention: Q*K^T --------------------------------
// S[g,h,i,j] = (1/sqrt(HD)) * sum_d q[g*PN+i, h*HD+d] * k[g*PN+j, h*HD+d]
__global__ __launch_bounds__(256) void k_attn_scores(
    const float* __restrict__ qkv, float* __restrict__ S)
{
    int lane = threadIdx.x & 31;
    int wave = threadIdx.x >> 5;
    int gh = blockIdx.y;                 // g*NH + h
    int g = gh >> 2, h = gh & 3;
    int tile = blockIdx.x * 8 + wave;    // 0..255 -> 16x16 grid of 16x16 tiles
    int i_base = (tile >> 4) * 16;
    int j_base = (tile & 15) * 16;

    const float* Q  = qkv + (size_t)(g * PN) * (3 * DM) + h * HD;
    const float* Kp = qkv + (size_t)(g * PN) * (3 * DM) + DM + h * HD;

    int irow  = i_base + (lane & 15);
    int lo    = (lane >> 4) * 16;
    int jcol  = j_base + (lane & 15);
    int koffB = (lane >> 4) * 16;
    const float* Arow = Q  + (size_t)irow * (3 * DM);
    const float* Brow = Kp + (size_t)jcol * (3 * DM) + koffB;

    v8f acc = {};
    #pragma unroll
    for (int k = 0; k < HD; k += 32) {
        v16h a, b;
        const float* pa = Arow + k + lo;
        #pragma unroll
        for (int i = 0; i < 8; ++i) {
            a[i]     = (_Float16)pa[i];
            a[8 + i] = (_Float16)pa[8 + i];
        }
        const float* pb = Brow + k;
        #pragma unroll
        for (int i = 0; i < 16; ++i) b[i] = (_Float16)pb[i];
        acc = __builtin_amdgcn_wmma_f32_16x16x32_f16(
                  false, a, false, b, (short)0, acc, false, false);
    }

    const float scale = 0.08838834764831845f;   // 1/sqrt(128)
    int mo = (lane >> 4) * 8;
    #pragma unroll
    for (int r = 0; r < 8; ++r) {
        int i = i_base + mo + r;
        S[((size_t)gh * PN + i) * PN + jcol] = acc[r] * scale;
    }
}

// row softmax over PN=256 entries (block = one row), LDS tree reduction
__global__ __launch_bounds__(256) void k_softmax256(float* __restrict__ S)
{
    __shared__ float red[256];
    int tid = threadIdx.x;
    size_t row = blockIdx.x;
    float v = S[row * PN + tid];
    red[tid] = v; __syncthreads();
    for (int s = 128; s > 0; s >>= 1) {
        if (tid < s) red[tid] = fmaxf(red[tid], red[tid + s]);
        __syncthreads();
    }
    float mx = red[0]; __syncthreads();
    float e = __expf(v - mx);
    red[tid] = e; __syncthreads();
    for (int s = 128; s > 0; s >>= 1) {
        if (tid < s) red[tid] += red[tid + s];
        __syncthreads();
    }
    S[row * PN + tid] = e / red[0];
}

// V^T staging: Vt[c, n] (f16, c = h*HD+d contiguous over nodes)
__global__ __launch_bounds__(256) void k_vt(
    const float* __restrict__ qkv, _Float16* __restrict__ Vt)
{
    int idx = blockIdx.x * 256 + threadIdx.x;   // NN*DM
    int n = idx >> 9;
    int c = idx & (DM - 1);
    Vt[(size_t)c * NN + n] = (_Float16)qkv[(size_t)n * (3 * DM) + 2 * DM + c];
}

// O[g*PN+i, h*HD+d] = sum_j S[g,h,i,j] * V[g*PN+j, h*HD+d]
__global__ __launch_bounds__(256) void k_attn_av(
    const float* __restrict__ S, const _Float16* __restrict__ Vt,
    float* __restrict__ O)
{
    int lane = threadIdx.x & 31;
    int wave = threadIdx.x >> 5;
    int gh = blockIdx.y;
    int g = gh >> 2, h = gh & 3;
    int tile = blockIdx.x * 8 + wave;   // 0..127 -> 16(i) x 8(d) tiles
    int i_base = (tile >> 3) * 16;
    int d_base = (tile & 7) * 16;

    const float* A = S + (size_t)gh * (PN * PN);
    int irow  = i_base + (lane & 15);
    int lo    = (lane >> 4) * 16;
    int dcol  = d_base + (lane & 15);
    int koffB = (lane >> 4) * 16;
    const float*    Arow = A + (size_t)irow * PN;
    const _Float16* Brow = Vt + (size_t)(h * HD + dcol) * NN + g * PN + koffB;

    v8f acc = {};
    for (int k = 0; k < PN; k += 32) {
        v16h a, b;
        const float* pa = Arow + k + lo;
        #pragma unroll
        for (int i = 0; i < 8; ++i) {
            a[i]     = (_Float16)pa[i];
            a[8 + i] = (_Float16)pa[8 + i];
        }
        b = *(const v16h*)(Brow + k);
        acc = __builtin_amdgcn_wmma_f32_16x16x32_f16(
                  false, a, false, b, (short)0, acc, false, false);
    }

    int mo = (lane >> 4) * 8;
    #pragma unroll
    for (int r = 0; r < 8; ++r) {
        int i = i_base + mo + r;
        O[(size_t)(g * PN + i) * DM + h * HD + dcol] = acc[r];
    }
}

// ------------------------------ small kernels ------------------------------
__global__ __launch_bounds__(256) void k_cvt(
    const float* __restrict__ src, _Float16* __restrict__ dst, int n)
{
    int i = blockIdx.x * 256 + threadIdx.x;
    if (i < n) dst[i] = (_Float16)src[i];
}

// W[K][N] -> Wt[N][K] f16
__global__ __launch_bounds__(256) void k_cvt_t(
    const float* __restrict__ W, _Float16* __restrict__ Wt, int K, int N)
{
    int idx = blockIdx.x * 256 + threadIdx.x;
    if (idx >= K * N) return;
    int k = idx / N, n = idx % N;
    Wt[(size_t)n * K + k] = (_Float16)W[idx];
}

__global__ __launch_bounds__(256) void k_fill(float* __restrict__ p, int n, float v)
{
    int i = blockIdx.x * 256 + threadIdx.x;
    if (i < n) p[i] = v;
}

// concat = [relu(pos @ W_pos + b_pos) | relu(emb[id])]   -> [NN, 2*DM]
__global__ __launch_bounds__(256) void k_embed(
    const float* __restrict__ pos, const int* __restrict__ ids,
    const float* __restrict__ Wp, const float* __restrict__ bp,
    const float* __restrict__ emb, float* __restrict__ concat)
{
    int idx = blockIdx.x * 256 + threadIdx.x;   // NN * 2*DM
    int n = idx >> 10;
    int d = idx & 1023;
    float v;
    if (d < DM) {
        v = bp[d] + pos[n * 3 + 0] * Wp[d]
                  + pos[n * 3 + 1] * Wp[DM + d]
                  + pos[n * 3 + 2] * Wp[2 * DM + d];
    } else {
        v = emb[(size_t)ids[n] * DM + (d - DM)];
    }
    concat[idx] = fmaxf(v, 0.0f);
}

// per-graph restarting sinusoidal PE added to node features
__global__ __launch_bounds__(256) void k_add_pe(float* __restrict__ node)
{
    int idx = blockIdx.x * 256 + threadIdx.x;   // NN*DM
    int n = idx >> 9;
    int d = idx & (DM - 1);
    int p = n & (PN - 1);                       // n - start[batch[n]] == n % 256
    float rate = __powf(10000.0f, -(float)(2 * (d >> 1)) / (float)DM);
    float ang = (float)p * rate;
    node[idx] += (d & 1) ? __cosf(ang) : __sinf(ang);
}

// g[b, coloff + d] = max over nodes of graph b
__global__ __launch_bounds__(256) void k_segmax(
    const float* __restrict__ src, float* __restrict__ g, int coloff)
{
    int b = blockIdx.x;
    int d = blockIdx.y * 256 + threadIdx.x;     // 0..DM-1
    float m = -INFINITY;
    for (int i = 0; i < PN; ++i)
        m = fmaxf(m, src[(size_t)(b * PN + i) * DM + d]);
    g[(size_t)b * ((1 + NL) * DM) + coloff + d] = m;
}

__global__ __launch_bounds__(256) void k_deg(
    const int* __restrict__ col, float* __restrict__ deg, int E)
{
    int e = blockIdx.x * 256 + threadIdx.x;
    if (e < E) atomicAdd(&deg[col[e]], 1.0f);
}

// acc[col] += rsqrt(deg[row])*rsqrt(deg[col]) * xw[row]
__global__ __launch_bounds__(128) void k_gcn_scatter(
    const int* __restrict__ row, const int* __restrict__ col,
    const float* __restrict__ deg, const float* __restrict__ xw,
    float* __restrict__ acc)
{
    int e = blockIdx.x;
    int t = threadIdx.x;
    int r = row[e], c = col[e];
    float coef = rsqrtf(deg[r]) * rsqrtf(deg[c]);
    const float* x = xw + (size_t)r * DM;
    float* a = acc + (size_t)c * DM;
    #pragma unroll
    for (int i = 0; i < 4; ++i) {
        int d = t + i * 128;
        atomicAdd(&a[d], coef * x[d]);
    }
}

// h = relu(acc + xw/deg + b)
__global__ __launch_bounds__(256) void k_gcn_final(
    const float* __restrict__ acc, const float* __restrict__ xw,
    const float* __restrict__ deg, const float* __restrict__ b,
    float* __restrict__ h)
{
    int idx = blockIdx.x * 256 + threadIdx.x;   // NN*DM
    int n = idx >> 9;
    int d = idx & (DM - 1);
    float v = acc[idx] + xw[idx] / deg[n] + b[d];
    h[idx] = fmaxf(v, 0.0f);
}

// ------------------------------ host launcher ------------------------------
extern "C" void kernel_launch(void* const* d_in, const int* in_sizes, int n_in,
                              void* d_out, int out_size, void* d_ws, size_t ws_size,
                              hipStream_t stream)
{
    (void)n_in; (void)out_size; (void)ws_size;
    const float* pos    = (const float*)d_in[0];
    const int*   ids    = (const int*)  d_in[1];
    const int*   eidx   = (const int*)  d_in[2];
    // d_in[3] = batch (implicit: n / PN)
    const float* W_pos  = (const float*)d_in[4];
    const float* b_pos  = (const float*)d_in[5];
    const float* emb    = (const float*)d_in[6];
    const float* W_node = (const float*)d_in[7];
    const float* b_node = (const float*)d_in[8];
    const float* gcn_W  = (const float*)d_in[9];
    const float* gcn_b  = (const float*)d_in[10];
    const float* in_w   = (const float*)d_in[11];
    const float* in_b   = (const float*)d_in[12];
    const float* out_w  = (const float*)d_in[13];
    const float* out_b  = (const float*)d_in[14];
    const float* lin_w  = (const float*)d_in[15];
    const float* lin_b  = (const float*)d_in[16];
    const float* W_agg  = (const float*)d_in[17];
    const float* b_agg  = (const float*)d_in[18];
    float* outp = (float*)d_out;
    const int E = in_sizes[2] / 2;

    char* ws = (char*)d_ws;
    size_t off = 0;
    auto alloc_f = [&](size_t n) {
        float* p = (float*)(ws + off);
        off = (off + n * 4 + 255) & ~(size_t)255;
        return p;
    };
    auto alloc_h = [&](size_t n) {
        _Float16* p = (_Float16*)(ws + off);
        off = (off + n * 2 + 255) & ~(size_t)255;
        return p;
    };

    float* u      = alloc_f((size_t)NN * 3 * DM);   // concat [NN,2DM] / qkv [NN,3DM]
    float* node   = alloc_f((size_t)NN * DM);
    float* hbuf   = alloc_f((size_t)NN * DM);
    float* xw     = alloc_f((size_t)NN * DM);
    float* acc    = alloc_f((size_t)NN * DM);
    float* S      = alloc_f((size_t)NB * NH * PN * PN);
    float* attno  = alloc_f((size_t)NN * DM);
    float* tmp    = alloc_f((size_t)NN * DM);
    float* deg    = alloc_f(NN);
    float* gbuf   = alloc_f((size_t)NB * (1 + NL) * DM);
    _Float16* Wt_node = alloc_h((size_t)DM * 2 * DM);
    _Float16* Wt_gcn  = alloc_h((size_t)NL * DM * DM);
    _Float16* Wh_in   = alloc_h((size_t)NL * 3 * DM * DM);
    _Float16* Wh_out  = alloc_h((size_t)NL * DM * DM);
    _Float16* Wh_lin  = alloc_h((size_t)NL * DM * DM);
    _Float16* Wt_agg  = alloc_h((size_t)DM * (1 + NL) * DM);
    _Float16* Vt      = alloc_h((size_t)DM * NN);

    auto gemm = [&](const float* A, const _Float16* Wt, const float* bias,
                    float* C, int M, int N, int K, int lda, int ldb, int ldc,
                    int relu) {
        dim3 grid((M + 31) / 32, N / 128);
        k_gemm_wmma<<<grid, 256, 0, stream>>>(A, Wt, bias, C, M, N, K,
                                              lda, ldb, ldc, relu);
    };

    // ---- weight prep (f16, [N][K]) ----
    k_cvt_t<<<(2 * DM * DM + 255) / 256, 256, 0, stream>>>(W_node, Wt_node, 2 * DM, DM);
    for (int l = 0; l < NL; ++l)
        k_cvt_t<<<(DM * DM + 255) / 256, 256, 0, stream>>>(
            gcn_W + (size_t)l * DM * DM, Wt_gcn + (size_t)l * DM * DM, DM, DM);
    k_cvt<<<(NL * 3 * DM * DM + 255) / 256, 256, 0, stream>>>(in_w, Wh_in, NL * 3 * DM * DM);
    k_cvt<<<(NL * DM * DM + 255) / 256, 256, 0, stream>>>(out_w, Wh_out, NL * DM * DM);
    k_cvt<<<(NL * DM * DM + 255) / 256, 256, 0, stream>>>(lin_w, Wh_lin, NL * DM * DM);
    k_cvt_t<<<((1 + NL) * DM * DM + 255) / 256, 256, 0, stream>>>(
        W_agg, Wt_agg, (1 + NL) * DM, DM);

    // ---- node features ----
    k_embed<<<(NN * 2 * DM) / 256, 256, 0, stream>>>(pos, ids, W_pos, b_pos, emb, u);
    gemm(u, Wt_node, b_node, node, NN, DM, 2 * DM, 2 * DM, 2 * DM, DM, 1);
    k_add_pe<<<(NN * DM) / 256, 256, 0, stream>>>(node);

    // ---- degrees (deg = in-degree + 1) ----
    k_fill<<<(NN + 255) / 256, 256, 0, stream>>>(deg, NN, 1.0f);
    k_deg<<<(E + 255) / 256, 256, 0, stream>>>(eidx + E, deg, E);

    // ---- readout init ----
    k_segmax<<<dim3(NB, DM / 256), 256, 0, stream>>>(node, gbuf, 0);

    float* cur = node;
    for (int l = 0; l < NL; ++l) {
        // GCN
        gemm(cur, Wt_gcn + (size_t)l * DM * DM, nullptr, xw, NN, DM, DM, DM, DM, DM, 0);
        k_fill<<<(NN * DM) / 256, 256, 0, stream>>>(acc, NN * DM, 0.0f);
        k_gcn_scatter<<<E, 128, 0, stream>>>(eidx, eidx + E, deg, xw, acc);
        k_gcn_final<<<(NN * DM) / 256, 256, 0, stream>>>(
            acc, xw, deg, gcn_b + (size_t)l * DM, hbuf);

        // masked multi-head attention (block-diagonal == per-graph)
        gemm(hbuf, Wh_in + (size_t)l * 3 * DM * DM, in_b + (size_t)l * 3 * DM,
             u, NN, 3 * DM, DM, DM, DM, 3 * DM, 0);
        k_vt<<<(NN * DM) / 256, 256, 0, stream>>>(u, Vt);
        k_attn_scores<<<dim3(32, NB * NH), 256, 0, stream>>>(u, S);
        k_softmax256<<<NB * NH * PN, 256, 0, stream>>>(S);
        k_attn_av<<<dim3(16, NB * NH), 256, 0, stream>>>(S, Vt, attno);
        gemm(attno, Wh_out + (size_t)l * DM * DM, out_b + (size_t)l * DM,
             tmp, NN, DM, DM, DM, DM, DM, 0);
        gemm(tmp, Wh_lin + (size_t)l * DM * DM, lin_b + (size_t)l * DM,
             cur, NN, DM, DM, DM, DM, DM, 1);   // relu -> layer output

        k_segmax<<<dim3(NB, DM / 256), 256, 0, stream>>>(cur, gbuf, DM * (l + 1));
        // cur stays: layer output written back into `node` buffer each layer
    }

    // ---- final projection: [8, 2048] x [2048, 512] ----
    gemm(gbuf, Wt_agg, b_agg, outp, NB, DM, (1 + NL) * DM,
         (1 + NL) * DM, (1 + NL) * DM, DM, 0);
}
